// Jammer_21234318311696
// MI455X (gfx1250) — matrix operand order repaired and checked by reading, not compiled, not run on hardware
//
#include <hip/hip_runtime.h>

typedef _Float16 f16;
typedef __attribute__((ext_vector_type(16))) _Float16 v16h;
typedef __attribute__((ext_vector_type(8)))  _Float16 f16x8;
typedef __attribute__((ext_vector_type(8)))  float    v8f;
typedef __attribute__((ext_vector_type(4)))  float    f32x4;

#define B_     8
#define SQ     2048
#define SK     2048
#define DIN    512
#define U_     200
#define UP     224        // padded units: 7 chunks of 32, 14 tiles of 16
#define NCHUNK 7          // UP/32
#define NT_OUT 13         // 13*16 = 208 >= 200 output u-tiles
#define UPAD   208        // NT_OUT*16, stride of partial-O rows
#define NROWS  16384      // B_*SQ

// ---- WMMA fragment loaders (wave32, layouts per ISA 7.12.2) -------------

// A fragment 16x32 f16 from row-major src (K contiguous within a row).
// lane 0-15: M=lane, K = 0..7 & 16..23 ; lane 16-31: M=lane-16, K = 8..15 & 24..31
static __device__ inline v16h loadA_f16(const f16* base, int stride) {
  int lane = threadIdx.x & 31;
  int m  = lane & 15;
  int kb = (lane & 16) ? 8 : 0;
  const f16* p = base + m * stride + kb;
  f16x8 lo = *(const f16x8*)p;
  f16x8 hi = *(const f16x8*)(p + 16);
  v16h a;
#pragma unroll
  for (int i = 0; i < 8; ++i) { a[i] = lo[i]; a[i + 8] = hi[i]; }
  return a;
}

// Same A fragment but source is f32 row-major (convert to f16).
static __device__ inline v16h loadA_f32(const float* __restrict__ base, int stride) {
  int lane = threadIdx.x & 31;
  int m  = lane & 15;
  int kb = (lane & 16) ? 8 : 0;
  const float* p = base + m * stride + kb;
  f32x4 a0 = *(const f32x4*)(p);
  f32x4 a1 = *(const f32x4*)(p + 4);
  f32x4 a2 = *(const f32x4*)(p + 16);
  f32x4 a3 = *(const f32x4*)(p + 20);
  v16h a;
#pragma unroll
  for (int i = 0; i < 4; ++i) {
    a[i]      = (f16)a0[i];
    a[i + 4]  = (f16)a1[i];
    a[i + 8]  = (f16)a2[i];
    a[i + 12] = (f16)a3[i];
  }
  return a;
}

// B fragment 32x16 (KxN) f16 from an "N-major" buffer: element (k,n) at src[n*stride + k].
// lane 0-15: N=lane, K = 0..7 & 16..23 ; lane 16-31: N=lane-16, K = 8..15 & 24..31
static __device__ inline v16h loadB_f16(const f16* nmajor, int stride) {
  int lane = threadIdx.x & 31;
  int n  = lane & 15;
  int kb = (lane & 16) ? 8 : 0;
  const f16* p = nmajor + n * stride + kb;
  f16x8 lo = *(const f16x8*)p;
  f16x8 hi = *(const f16x8*)(p + 16);
  v16h b;
#pragma unroll
  for (int i = 0; i < 8; ++i) { b[i] = lo[i]; b[i + 8] = hi[i]; }
  return b;
}

// Async global->LDS 16-byte transfer (CDNA5 GLOBAL_LOAD_ASYNC_TO_LDS_B128, ASYNCcnt).
static __device__ inline void async_copy16(unsigned lds_off, const void* gaddr) {
  asm volatile("global_load_async_to_lds_b128 %0, %1, off"
               :: "v"(lds_off), "v"(gaddr) : "memory");
}
static __device__ inline void wait_async0() {
  asm volatile("s_wait_asynccnt 0x0" ::: "memory");
}

// ---- Kernel 1: weight transpose/convert + bias padding -------------------
__global__ void prep_kernel(const float* __restrict__ Wq, const float* __restrict__ Wk,
                            const float* __restrict__ Wv, const float* __restrict__ bq,
                            const float* __restrict__ bk, const float* __restrict__ bv,
                            f16* __restrict__ WT, float* __restrict__ biasP) {
  int idx = blockIdx.x * blockDim.x + threadIdx.x;
  const int total = 3 * UP * DIN;
  if (idx < total) {
    int p = idx / (UP * DIN);
    int r = idx % (UP * DIN);
    int u = r / DIN;
    int d = r % DIN;
    const float* W = (p == 0) ? Wq : (p == 1) ? Wk : Wv;
    WT[idx] = (u < U_) ? (f16)W[d * U_ + u] : (f16)0.f;
  }
  if (idx < 3 * UP) {
    int p = idx / UP;
    int u = idx % UP;
    const float* bias = (p == 0) ? bq : (p == 1) ? bk : bv;
    biasP[idx] = (u < U_) ? bias[u] : 0.f;
  }
}

// ---- Kernel 2: fused q/k/v projections via WMMA --------------------------
__global__ void __launch_bounds__(128)
proj_kernel(const float* __restrict__ gen, const float* __restrict__ real,
            const f16* __restrict__ WT, const float* __restrict__ biasP,
            f16* __restrict__ qbuf, f16* __restrict__ kbuf, f16* __restrict__ vT) {
  int p = blockIdx.y;
  const float* X   = (p == 0) ? gen : real;
  const f16* W     = WT + p * UP * DIN;
  const float* bia = biasP + p * UP;

  int wave = threadIdx.x >> 5;
  int lane = threadIdx.x & 31;
  int tile = blockIdx.x * 4 + wave;   // 0..1023 (16384 rows / 16)
  int row0 = tile * 16;
  int n    = lane & 15;
  int half = (lane & 16) ? 8 : 0;

#pragma unroll 1
  for (int nt = 0; nt < UP / 16; ++nt) {
    int n0 = nt * 16;
    v8f c = {};
#pragma unroll 1
    for (int kc = 0; kc < DIN / 32; ++kc) {
      v16h a = loadA_f32(X + (size_t)row0 * DIN + kc * 32, DIN);
      v16h b = loadB_f16(W + (size_t)n0 * DIN + kc * 32, DIN);
      c = __builtin_amdgcn_wmma_f32_16x16x32_f16(false, a, false, b, (short)0, c, false, false);
    }
    float bv = bia[n0 + n];  // zero in padding -> padded cols stay 0
    if (p < 2) {
      f16* dst = (p == 0) ? qbuf : kbuf;
#pragma unroll
      for (int j = 0; j < 8; ++j) {
        int row = row0 + half + j;   // C layout: VGPR j -> M = j (+8 for hi half)
        dst[(size_t)row * UP + n0 + n] = (f16)(c[j] + bv);
      }
    } else {
      // vT[b][u][t]: M (=t) contiguous across C VGPRs -> single packed b128 store
      int bidx = row0 / SK;
      int t0   = (row0 % SK) + half;
      f16x8 pk;
#pragma unroll
      for (int j = 0; j < 8; ++j) pk[j] = (f16)(c[j] + bv);
      *(f16x8*)(vT + ((size_t)bidx * UP + (n0 + n)) * SK + t0) = pk;
    }
  }
}

// ---- Kernel 3: flash attention, split-K, block-shared k/v tiles in LDS ---
// Block = 4 adjacent query tiles x SAME key split (blockIdx.y) -> all 4 waves
// consume identical k/v tiles; stage them once per block via async-to-LDS.
__global__ void __launch_bounds__(128)
attn_kernel(const f16* __restrict__ qbuf, const f16* __restrict__ kbuf,
            const f16* __restrict__ vT, float* __restrict__ Opart,
            float* __restrict__ Mstat, float* __restrict__ Lstat,
            int ksLog2) {
  __shared__ __align__(16) f16 kS[32 * UP];      // 14336 B: keys t0..t0+31, row-major [t][u]
  __shared__ __align__(16) f16 vS[UPAD * 32];    // 13312 B: [u][t] slices of vT
  __shared__ __align__(16) f16 Plds[4][16 * 32]; //  4096 B: per-wave P repack

  int tid  = threadIdx.x;
  int wave = tid >> 5;
  int lane = tid & 31;
  int split = blockIdx.y;
  int qt   = blockIdx.x * 4 + wave;     // 0..1023 (128 tiles/batch; 4 | 128 -> same batch)
  int b    = qt >> 7;
  int gr0  = qt * 16;
  int nkt  = (SK / 32) >> ksLog2;
  int ktBeg = split * nkt;
  const f16* kb_ = kbuf + (size_t)b * SK * UP;
  const f16* vb_ = vT   + (size_t)b * UP * SK;
  int n    = lane & 15;
  int half = (lane & 16) ? 8 : 0;
  const float scale = 0.07071067811865475f;  // 1/sqrt(200)

  unsigned kS0 = (unsigned)(size_t)&kS[0];
  unsigned vS0 = (unsigned)(size_t)&vS[0];

  v8f O[NT_OUT];
#pragma unroll
  for (int t = 0; t < NT_OUT; ++t) O[t] = (v8f){};
  float mi[8], li[8];
#pragma unroll
  for (int j = 0; j < 8; ++j) { mi[j] = -1e30f; li[j] = 0.f; }

  f16* myP = &Plds[wave][0];

  for (int kt = ktBeg; kt < ktBeg + nkt; ++kt) {
    int t0 = kt * 32;
    // ---- cooperative async staging of k/v tiles --------------------------
    __syncthreads();  // previous iteration's LDS reads complete
    {
      // k tile: contiguous 14336 B (32 key rows x 224 f16)
      const char* ksrc = (const char*)(kb_ + (size_t)t0 * UP);
#pragma unroll
      for (int i = 0; i < 7; ++i) {
        int idx = tid + i * 128;               // 896 x 16B
        async_copy16(kS0 + idx * 16, ksrc + idx * 16);
      }
      // v tile: 208 u-rows x 64 B (vT row stride = SK)
#pragma unroll
      for (int i = 0; i < 7; ++i) {
        int idx = tid + i * 128;               // 832 x 16B
        if (idx < UPAD * 4) {
          int u = idx >> 2, c = idx & 3;
          async_copy16(vS0 + u * 64 + c * 16,
                       (const char*)(vb_ + (size_t)u * SK + t0) + c * 16);
        }
      }
    }
    wait_async0();
    __syncthreads();  // staged tiles visible to all waves

    // ---- S = q k^T over padded U (7 chunks of 32) ------------------------
    v8f s0 = {}, s1 = {};
#pragma unroll 1
    for (int uc = 0; uc < NCHUNK; ++uc) {
      v16h a   = loadA_f16(qbuf + (size_t)gr0 * UP + uc * 32, UP);
      v16h bk0 = loadB_f16(&kS[0 * UP] + uc * 32, UP);
      s0 = __builtin_amdgcn_wmma_f32_16x16x32_f16(false, a, false, bk0, (short)0, s0, false, false);
      v16h bk1 = loadB_f16(&kS[16 * UP] + uc * 32, UP);
      s1 = __builtin_amdgcn_wmma_f32_16x16x32_f16(false, a, false, bk1, (short)0, s1, false, false);
    }
    // ---- online softmax (row m in VGPR j across one 16-lane half) --------
    float p0a[8], p1a[8], alpha[8];
#pragma unroll
    for (int j = 0; j < 8; ++j) {
      float a0 = s0[j] * scale, a1 = s1[j] * scale;
      float tm = fmaxf(a0, a1);
      tm = fmaxf(tm, __shfl_xor(tm, 1, 16));
      tm = fmaxf(tm, __shfl_xor(tm, 2, 16));
      tm = fmaxf(tm, __shfl_xor(tm, 4, 16));
      tm = fmaxf(tm, __shfl_xor(tm, 8, 16));
      float mn = fmaxf(mi[j], tm);
      float al = __expf(mi[j] - mn);
      mi[j] = mn;
      float p0 = __expf(a0 - mn);
      float p1 = __expf(a1 - mn);
      float rs = p0 + p1;
      rs += __shfl_xor(rs, 1, 16);
      rs += __shfl_xor(rs, 2, 16);
      rs += __shfl_xor(rs, 4, 16);
      rs += __shfl_xor(rs, 8, 16);
      li[j] = li[j] * al + rs;
      alpha[j] = al;
      p0a[j] = p0; p1a[j] = p1;
    }
#pragma unroll
    for (int t = 0; t < NT_OUT; ++t)
#pragma unroll
      for (int j = 0; j < 8; ++j) O[t][j] *= alpha[j];

    // ---- P repack (C layout f32 -> A layout f16) via per-wave LDS tile ---
#pragma unroll
    for (int j = 0; j < 8; ++j) {
      int m = half + j;
      myP[m * 32 + n]      = (f16)p0a[j];
      myP[m * 32 + 16 + n] = (f16)p1a[j];
    }
    __builtin_amdgcn_wave_barrier();  // per-wave tile: order LDS writes/reads
    v16h pf = loadA_f16(myP, 32);
    // ---- O += P V from staged v tile -------------------------------------
#pragma unroll
    for (int ut = 0; ut < NT_OUT; ++ut) {
      v16h vv = loadB_f16(&vS[(ut * 16) * 32], 32);
      O[ut] = __builtin_amdgcn_wmma_f32_16x16x32_f16(false, pf, false, vv, (short)0, O[ut], false, false);
    }
  }

  // Emit unnormalized partial O and per-row stats for the combine pass.
  float* op = Opart + (size_t)split * NROWS * UPAD;
#pragma unroll
  for (int ut = 0; ut < NT_OUT; ++ut) {
    int u = ut * 16 + n;
#pragma unroll
    for (int j = 0; j < 8; ++j) {
      int row = gr0 + half + j;
      op[(size_t)row * UPAD + u] = O[ut][j];
    }
  }
  if ((lane & 15) == 0) {  // stats replicated across each 16-lane half
#pragma unroll
    for (int j = 0; j < 8; ++j) {
      int row = gr0 + half + j;
      Mstat[(size_t)split * NROWS + row] = mi[j];
      Lstat[(size_t)split * NROWS + row] = li[j];
    }
  }
}

// ---- Kernel 4: combine split-K partials ----------------------------------
__global__ void combine_kernel(const float* __restrict__ Opart,
                               const float* __restrict__ Mstat,
                               const float* __restrict__ Lstat,
                               float* __restrict__ out, int ks) {
  int idx = blockIdx.x * blockDim.x + threadIdx.x;
  if (idx >= NROWS * U_) return;
  int row = idx / U_;
  int u   = idx % U_;
  float mmax = -1e30f;
  for (int s = 0; s < ks; ++s) mmax = fmaxf(mmax, Mstat[(size_t)s * NROWS + row]);
  float denom = 0.f, num = 0.f;
  for (int s = 0; s < ks; ++s) {
    float wgt = __expf(Mstat[(size_t)s * NROWS + row] - mmax);
    denom += Lstat[(size_t)s * NROWS + row] * wgt;
    num   += Opart[((size_t)s * NROWS + row) * UPAD + u] * wgt;
  }
  out[idx] = num / denom;
}

// ---- Host launch ---------------------------------------------------------
extern "C" void kernel_launch(void* const* d_in, const int* in_sizes, int n_in,
                              void* d_out, int out_size, void* d_ws, size_t ws_size,
                              hipStream_t stream) {
  const float* gen  = (const float*)d_in[0];
  const float* real = (const float*)d_in[1];
  const float* Wq   = (const float*)d_in[2];
  const float* bq   = (const float*)d_in[3];
  const float* Wk   = (const float*)d_in[4];
  const float* bk   = (const float*)d_in[5];
  const float* Wv   = (const float*)d_in[6];
  const float* bv   = (const float*)d_in[7];
  float* out = (float*)d_out;

  char* ws = (char*)d_ws;
  f16*   WT    = (f16*)(ws);                 //   688,128 B
  float* biasP = (float*)(ws + 688128);      //     2,688 B
  f16*   qbuf  = (f16*)(ws + 692224);        // 7,340,032 B
  f16*   kbuf  = (f16*)(ws + 8032256);       // 7,340,032 B
  f16*   vTb   = (f16*)(ws + 15372288);      // 7,340,032 B
  float* Mstat = (float*)(ws + 22712320);    // 4*16384*4 = 262,144 B
  float* Lstat = (float*)(ws + 22974464);    //             262,144 B
  float* Opart = (float*)(ws + 23236608);    // ks * 16384*208*4 (13,631,488 B per split)

  // Key-split from available workspace (deterministic in inputs):
  // ks=4 -> 4096 waves (~4/SIMD on an MI455X-scale part), partials L2-resident.
  const size_t base = 23236608;
  const size_t perSplit = (size_t)NROWS * UPAD * 4;
  int ksLog2 = (ws_size >= base + 4 * perSplit) ? 2
             : (ws_size >= base + 2 * perSplit) ? 1 : 0;
  int ks = 1 << ksLog2;

  int prepN = 3 * UP * DIN;
  prep_kernel<<<(prepN + 255) / 256, 256, 0, stream>>>(Wq, Wk, Wv, bq, bk, bv, WT, biasP);
  // 16384 rows / 16 per wave / 4 waves per block = 256 blocks; y = {q,k,v}
  proj_kernel<<<dim3(256, 3), 128, 0, stream>>>(gen, real, WT, biasP, qbuf, kbuf, vTb);
  // x: 256 groups of 4 query tiles; y: key split
  attn_kernel<<<dim3(256, ks), 128, 0, stream>>>(qbuf, kbuf, vTb, Opart, Mstat, Lstat, ksLog2);
  combine_kernel<<<(NROWS * U_ + 255) / 256, 256, 0, stream>>>(Opart, Mstat, Lstat, out, ks);
}
